// DeepseekV2AttentionMLA_53609781788758
// MI455X (gfx1250) — matrix operand order repaired and checked by reading, not compiled, not run on hardware
//
#include <hip/hip_runtime.h>
#include <stdint.h>

typedef __bf16 bf16;
typedef __bf16 v16bf __attribute__((ext_vector_type(16)));
typedef __bf16 v8bf  __attribute__((ext_vector_type(8)));
typedef float  v8f   __attribute__((ext_vector_type(8)));

#define N_TOK   2048
#define HID_D   5120
#define N_HEAD  16
#define D_NOPE  128
#define D_ROPE  64
#define D_VAL   128
#define Q_LORA  1536
#define KV_LORA 512
#define QKV_W   2112      /* Q_LORA + KV_LORA + D_ROPE */
#define QB_W    3072      /* N_HEAD*(D_NOPE+D_ROPE)    */
#define QFULL_W 576       /* KV_LORA + D_ROPE          */
#define ATT_SCALE 0.07216878364870323f   /* (128+64)^-0.5 */
#define LN_THETA  9.210340371976184f     /* ln(10000)     */
#define BM_GEMM 256       /* GEMM row block == causal band */

// ---- CDNA5 async global->LDS (ASYNCcnt), inline-asm form ------------------
#if defined(__AMDGCN__)
#define USE_ASYNC 1
#else
#define USE_ASYNC 0
#endif

#if USE_ASYNC
__device__ __forceinline__ void async_cp16(const bf16* g, bf16* l) {
  // vdst = LDS byte address (generic LDS ptr low 32 bits), vaddr = global addr
  asm volatile("global_load_async_to_lds_b128 %0, %1, off"
               :: "v"((uint32_t)(uintptr_t)l),
                  "v"((uint64_t)(uintptr_t)g)
               : "memory");
}
template<int N> __device__ __forceinline__ void wait_async() {
#if __has_builtin(__builtin_amdgcn_s_wait_asynccnt)
  __builtin_amdgcn_s_wait_asynccnt(N);
#else
  asm volatile("s_wait_asynccnt %0" :: "i"(N) : "memory");
#endif
}
#endif

// ---------------------------------------------------------------------------
// NN GEMM: C[M,N] = A[M,K](bf16 row) * B[K,N](bf16 row); BM=256, BN=NT*16,
// BK=32, 256 threads = 8 waves. Wave w owns rows [32w,32w+32): 2 m-tiles x NT
// n-tiles = 2*NT wmma(16x16x32 bf16) per K step. Double-buffered LDS fed by
// async-to-LDS loads (ASYNCcnt), one tile ahead.
// mode: 0 plain, 1 causal tile skip (scores), 2 K clipped to m0+BM (P*V)
// ---------------------------------------------------------------------------
template<int NT, bool OUT_BF16>
__global__ __launch_bounds__(256) void gemm_nn_kernel(
    const bf16* __restrict__ A, int lda,
    const bf16* __restrict__ B, int ldb,
    void* __restrict__ Cv, int ldc,
    int K, int mode,
    long strideA, long strideB, long strideC)
{
  constexpr int BM = BM_GEMM, BN = NT * 16, BK = 32;
  constexpr int LDA_S = 40, LDB_S = BN + 8;
  constexpr int BCH = NT / 4;            // B 16B-chunks per thread per tile
  constexpr int NASYNC = 4 + BCH;        // async ops per tile per wave
  __shared__ bf16 As[2][BM][LDA_S];
  __shared__ bf16 Bs[2][BK][LDB_S];

  const int m0 = blockIdx.y * BM;
  const int n0 = blockIdx.x * BN;
  if (mode == 1 && n0 >= m0 + BM) return;
  const int kEnd = (mode == 2) ? ((K < m0 + BM) ? K : (m0 + BM)) : K;

  A += (size_t)blockIdx.z * strideA;
  B += (size_t)blockIdx.z * strideB;

  const int tid  = threadIdx.x;
  const int wave = tid >> 5, lane = tid & 31;
  const int half = lane >> 4;
  const int mrow = (wave << 5) + (lane & 15);
  // loader geometry (16B chunks)
  const int ar = tid >> 2,          ac = (tid & 3) << 3;          // A: 256x32
  const int br = tid / (2 * NT),    bc = (tid % (2 * NT)) << 3;   // B: 32xBN

  v8f acc[2][NT] = {};

  auto do_compute = [&](int buf) {
    v16bf a0, a1;
    {
      v8bf lo = *(const v8bf*)&As[buf][mrow][half * 8];
      v8bf hi = *(const v8bf*)&As[buf][mrow][16 + half * 8];
      #pragma unroll
      for (int j = 0; j < 8; ++j) { a0[j] = lo[j]; a0[8 + j] = hi[j]; }
    }
    {
      v8bf lo = *(const v8bf*)&As[buf][mrow + 16][half * 8];
      v8bf hi = *(const v8bf*)&As[buf][mrow + 16][16 + half * 8];
      #pragma unroll
      for (int j = 0; j < 8; ++j) { a1[j] = lo[j]; a1[8 + j] = hi[j]; }
    }
    #pragma unroll
    for (int nt = 0; nt < NT; ++nt) {
      v16bf b;
      #pragma unroll
      for (int j = 0; j < 8; ++j) {
        b[j]     = Bs[buf][half * 8 + j][nt * 16 + (lane & 15)];
        b[8 + j] = Bs[buf][16 + half * 8 + j][nt * 16 + (lane & 15)];
      }
      acc[0][nt] = __builtin_amdgcn_wmma_f32_16x16x32_bf16(
          false, a0, false, b, (short)0, acc[0][nt], false, false);
      acc[1][nt] = __builtin_amdgcn_wmma_f32_16x16x32_bf16(
          false, a1, false, b, (short)0, acc[1][nt], false, false);
    }
  };

  const int nk = kEnd / BK;
#if USE_ASYNC
  auto issue = [&](int buf, int k0) {
    #pragma unroll
    for (int c = 0; c < 4; ++c)
      async_cp16(A + (size_t)(m0 + c * 64 + ar) * lda + (k0 + ac),
                 &As[buf][c * 64 + ar][ac]);
    #pragma unroll
    for (int c = 0; c < BCH; ++c)
      async_cp16(B + (size_t)(k0 + c * (BK / BCH) + br) * ldb + (n0 + bc),
                 &Bs[buf][c * (BK / BCH) + br][bc]);
  };
  issue(0, 0);
  for (int i = 0; i < nk; ++i) {
    const int cur = i & 1;
    if (i + 1 < nk) { issue(cur ^ 1, (i + 1) * BK); wait_async<NASYNC>(); }
    else            { wait_async<0>(); }
    __syncthreads();                 // tile i resident for everyone
    do_compute(cur);
    __syncthreads();                 // buffer cur free for re-issue
  }
#else
  for (int i = 0; i < nk; ++i) {
    const int k0 = i * BK;
    uint4 av[4], bv[BCH];
    #pragma unroll
    for (int c = 0; c < 4; ++c)
      av[c] = *(const uint4*)(A + (size_t)(m0 + c * 64 + ar) * lda + (k0 + ac));
    #pragma unroll
    for (int c = 0; c < BCH; ++c)
      bv[c] = *(const uint4*)(B + (size_t)(k0 + c * (BK / BCH) + br) * ldb + (n0 + bc));
    #pragma unroll
    for (int c = 0; c < 4; ++c) *(uint4*)&As[0][c * 64 + ar][ac] = av[c];
    #pragma unroll
    for (int c = 0; c < BCH; ++c) *(uint4*)&Bs[0][c * (BK / BCH) + br][bc] = bv[c];
    __syncthreads();
    do_compute(0);
    __syncthreads();
  }
#endif

  #pragma unroll
  for (int mt = 0; mt < 2; ++mt)
    #pragma unroll
    for (int nt = 0; nt < NT; ++nt)
      #pragma unroll
      for (int r = 0; r < 8; ++r) {
        const int row = m0 + (wave << 5) + (mt << 4) + (half << 3) + r;
        const int col = n0 + (nt << 4) + (lane & 15);
        if (OUT_BF16) {
          bf16* C = (bf16*)Cv + (size_t)blockIdx.z * strideC;
          C[(size_t)row * ldc + col] = (bf16)acc[mt][nt][r];
        } else {
          float* C = (float*)Cv + (size_t)blockIdx.z * strideC;
          C[(size_t)row * ldc + col] = acc[mt][nt][r];
        }
      }
}

// ---------------------------------------------------------------------------
__global__ __launch_bounds__(256) void cast_bf16_kernel(
    const float* __restrict__ in, bf16* __restrict__ out, long n)
{
  long i = ((long)blockIdx.x * 256 + threadIdx.x) * 4;
  if (i + 4 <= n) {
    float4 v = *(const float4*)(in + i);
    out[i]     = (bf16)v.x;
    out[i + 1] = (bf16)v.y;
    out[i + 2] = (bf16)v.z;
    out[i + 3] = (bf16)v.w;
  }
}

// w_kvb (512 x [16*256]) -> wkc[h][d=128][l=512], wvc[h][l=512][v=128]
__global__ __launch_bounds__(256) void split_wkvb_kernel(
    const float* __restrict__ wkvb, bf16* __restrict__ wkc, bf16* __restrict__ wvc)
{
  int idx = blockIdx.x * 256 + threadIdx.x;
  int d = idx & 255, h = (idx >> 8) & 15, l = idx >> 12;
  if (l >= KV_LORA) return;
  float v = wkvb[(size_t)l * 4096 + h * 256 + d];
  if (d < D_NOPE)
    wkc[((size_t)h * D_NOPE + d) * KV_LORA + l] = (bf16)v;
  else
    wvc[((size_t)h * KV_LORA + l) * D_VAL + (d - D_NOPE)] = (bf16)v;
}

__global__ __launch_bounds__(256) void rms_k_kernel(
    const float* __restrict__ qkv, const float* __restrict__ qa_w,
    const float* __restrict__ kva_w, const int* __restrict__ pos,
    bf16* __restrict__ q_an, bf16* __restrict__ kv_lat, bf16* __restrict__ k_fullT)
{
  __shared__ float red[256];
  const int t = blockIdx.x, tid = threadIdx.x;
  const float* row = qkv + (size_t)t * QKV_W;

  float ss = 0.f;
  for (int i = tid; i < Q_LORA; i += 256) { float v = row[i]; ss += v * v; }
  red[tid] = ss; __syncthreads();
  for (int s = 128; s > 0; s >>= 1) { if (tid < s) red[tid] += red[tid + s]; __syncthreads(); }
  const float rq = rsqrtf(red[0] / (float)Q_LORA + 1e-6f);
  __syncthreads();
  for (int i = tid; i < Q_LORA; i += 256)
    q_an[(size_t)t * Q_LORA + i] = (bf16)(row[i] * rq * qa_w[i]);

  ss = 0.f;
  for (int i = tid; i < KV_LORA; i += 256) { float v = row[Q_LORA + i]; ss += v * v; }
  red[tid] = ss; __syncthreads();
  for (int s = 128; s > 0; s >>= 1) { if (tid < s) red[tid] += red[tid + s]; __syncthreads(); }
  const float rkv = rsqrtf(red[0] / (float)KV_LORA + 1e-6f);
  __syncthreads();
  for (int i = tid; i < KV_LORA; i += 256) {
    float v = row[Q_LORA + i] * rkv * kva_w[i];
    kv_lat[(size_t)t * KV_LORA + i] = (bf16)v;
    k_fullT[(size_t)i * N_TOK + t]  = (bf16)v;
  }

  if (tid < D_ROPE / 2) {
    const int p = tid;
    const float fpos = (float)pos[t];
    const float inv  = __expf(-(float)p * (LN_THETA / 32.f));
    float s, c; sincosf(fpos * inv, &s, &c);
    const float x1 = row[Q_LORA + KV_LORA + 2 * p];
    const float x2 = row[Q_LORA + KV_LORA + 2 * p + 1];
    k_fullT[(size_t)(KV_LORA + 2 * p) * N_TOK + t]     = (bf16)(x1 * c - x2 * s);
    k_fullT[(size_t)(KV_LORA + 2 * p + 1) * N_TOK + t] = (bf16)(x2 * c + x1 * s);
  }
}

__global__ __launch_bounds__(256) void build_q_kernel(
    const float* __restrict__ qraw, const int* __restrict__ pos,
    bf16* __restrict__ q_nope, bf16* __restrict__ q_full)
{
  const int t = blockIdx.x, tid = threadIdx.x;
  const float* row = qraw + (size_t)t * QB_W;
  const float fpos = (float)pos[t];
  for (int P = tid; P < 1536; P += 256) {
    const int h = P / 96, k = P % 96;
    if (k < 64) {
      const int d = k * 2;
      bf16* dst = q_nope + ((size_t)h * N_TOK + t) * D_NOPE + d;
      dst[0] = (bf16)row[h * 192 + d];
      dst[1] = (bf16)row[h * 192 + d + 1];
    } else {
      const int p = k - 64;
      const float inv = __expf(-(float)p * (LN_THETA / 32.f));
      float s, c; sincosf(fpos * inv, &s, &c);
      const float x1 = row[h * 192 + D_NOPE + 2 * p];
      const float x2 = row[h * 192 + D_NOPE + 2 * p + 1];
      bf16* dst = q_full + ((size_t)h * N_TOK + t) * QFULL_W + KV_LORA + 2 * p;
      dst[0] = (bf16)(x1 * c - x2 * s);
      dst[1] = (bf16)(x2 * c + x1 * s);
    }
  }
}

// one wave per row; zero-pad P out to the 256-row causal band edge
__global__ __launch_bounds__(256) void softmax_kernel(
    const float* __restrict__ S, bf16* __restrict__ P)
{
  const int wave = threadIdx.x >> 5, lane = threadIdx.x & 31;
  const int t = blockIdx.x * 8 + wave;
  const float* srow = S + (size_t)t * N_TOK;
  bf16* prow = P + (size_t)t * N_TOK;

  float mx = -3.4e38f;
  for (int s = lane; s <= t; s += 32) mx = fmaxf(mx, srow[s] * ATT_SCALE);
  #pragma unroll
  for (int o = 16; o; o >>= 1) mx = fmaxf(mx, __shfl_xor(mx, o, 32));
  float sum = 0.f;
  for (int s = lane; s <= t; s += 32) sum += __expf(srow[s] * ATT_SCALE - mx);
  #pragma unroll
  for (int o = 16; o; o >>= 1) sum += __shfl_xor(sum, o, 32);
  const float inv = 1.f / sum;
  for (int s = lane; s <= t; s += 32)
    prow[s] = (bf16)(__expf(srow[s] * ATT_SCALE - mx) * inv);
  const int pad_end = ((t >> 8) + 1) << 8;   // BM_GEMM band edge
  for (int s = t + 1 + lane; s < pad_end; s += 32) prow[s] = (bf16)0.f;
}

// ---------------------------------------------------------------------------
static void launch_gemm(const bf16* A, int lda, const bf16* B, int ldb,
                        void* C, int ldc, int M, int N, int K, int mode,
                        bool outbf, int nt, long sA, long sB, long sC,
                        int batch, hipStream_t stream)
{
  dim3 grid(N / (nt * 16), M / BM_GEMM, batch), block(256);
  if (nt == 8) {
    if (outbf) gemm_nn_kernel<8, true ><<<grid, block, 0, stream>>>(A, lda, B, ldb, C, ldc, K, mode, sA, sB, sC);
    else       gemm_nn_kernel<8, false><<<grid, block, 0, stream>>>(A, lda, B, ldb, C, ldc, K, mode, sA, sB, sC);
  } else {
    if (outbf) gemm_nn_kernel<4, true ><<<grid, block, 0, stream>>>(A, lda, B, ldb, C, ldc, K, mode, sA, sB, sC);
    else       gemm_nn_kernel<4, false><<<grid, block, 0, stream>>>(A, lda, B, ldb, C, ldc, K, mode, sA, sB, sC);
  }
}

extern "C" void kernel_launch(void* const* d_in, const int* in_sizes, int n_in,
                              void* d_out, int out_size, void* d_ws, size_t ws_size,
                              hipStream_t stream)
{
  (void)in_sizes; (void)n_in; (void)out_size; (void)ws_size;
  const int*   positions = (const int*)d_in[0];
  const float* hs        = (const float*)d_in[1];
  const float* wfused    = (const float*)d_in[2];
  const float* wqb       = (const float*)d_in[3];
  const float* wkvb      = (const float*)d_in[4];
  const float* wo        = (const float*)d_in[5];
  const float* qa_w      = (const float*)d_in[6];
  const float* kva_w     = (const float*)d_in[7];

  char* p = (char*)d_ws;
  auto take = [&](size_t bytes) -> char* {
    char* r = p; p += (bytes + 255) & ~(size_t)255; return r;
  };
  bf16*  hs_b     = (bf16*) take((size_t)N_TOK * HID_D * 2);
  bf16*  wf_b     = (bf16*) take((size_t)HID_D * QKV_W * 2);
  bf16*  wqb_b    = (bf16*) take((size_t)Q_LORA * QB_W * 2);
  bf16*  wo_b     = (bf16*) take((size_t)N_HEAD * D_VAL * HID_D * 2);
  bf16*  wkc      = (bf16*) take((size_t)N_HEAD * D_NOPE * KV_LORA * 2);
  bf16*  wvc      = (bf16*) take((size_t)N_HEAD * KV_LORA * D_VAL * 2);
  float* qkv      = (float*)take((size_t)N_TOK * QKV_W * 4);
  bf16*  q_an     = (bf16*) take((size_t)N_TOK * Q_LORA * 2);
  bf16*  kv_lat   = (bf16*) take((size_t)N_TOK * KV_LORA * 2);
  bf16*  k_fullT  = (bf16*) take((size_t)QFULL_W * N_TOK * 2);
  float* qraw     = (float*)take((size_t)N_TOK * QB_W * 4);
  bf16*  q_nope   = (bf16*) take((size_t)N_HEAD * N_TOK * D_NOPE * 2);
  bf16*  q_full   = (bf16*) take((size_t)N_HEAD * N_TOK * QFULL_W * 2);
  float* scores   = (float*)take((size_t)N_TOK * N_TOK * 4);
  bf16*  Pm       = (bf16*) take((size_t)N_TOK * N_TOK * 2);
  bf16*  o_lat    = (bf16*) take((size_t)N_TOK * KV_LORA * 2);
  bf16*  o_cat    = (bf16*) take((size_t)N_TOK * N_HEAD * D_VAL * 2);

  auto cast = [&](const float* in, bf16* out, long n) {
    long blocks = (n / 4 + 255) / 256;
    cast_bf16_kernel<<<dim3((unsigned)blocks), dim3(256), 0, stream>>>(in, out, n);
  };
  cast(hs,     hs_b,  (long)N_TOK * HID_D);
  cast(wfused, wf_b,  (long)HID_D * QKV_W);
  cast(wqb,    wqb_b, (long)Q_LORA * QB_W);
  cast(wo,     wo_b,  (long)N_HEAD * D_VAL * HID_D);
  split_wkvb_kernel<<<dim3(8192), dim3(256), 0, stream>>>(wkvb, wkc, wvc);

  // qkv = hs @ w_fused : main 2048 cols (NT=8) + 64-col tail (NT=4)
  launch_gemm(hs_b, HID_D, wf_b, QKV_W, qkv, QKV_W,
              N_TOK, 2048, HID_D, 0, false, 8, 0, 0, 0, 1, stream);
  launch_gemm(hs_b, HID_D, wf_b + 2048, QKV_W, qkv + 2048, QKV_W,
              N_TOK, 64, HID_D, 0, false, 4, 0, 0, 0, 1, stream);

  rms_k_kernel<<<dim3(N_TOK), dim3(256), 0, stream>>>(
      qkv, qa_w, kva_w, positions, q_an, kv_lat, k_fullT);

  // q_raw = rmsnorm(q_a) @ w_qb
  launch_gemm(q_an, Q_LORA, wqb_b, QB_W, qraw, QB_W,
              N_TOK, QB_W, Q_LORA, 0, false, 8, 0, 0, 0, 1, stream);

  build_q_kernel<<<dim3(N_TOK), dim3(256), 0, stream>>>(qraw, positions, q_nope, q_full);

  // q_full[h][:, :512] = q_nope[h] @ w_kc[h]  (batched over heads)
  launch_gemm(q_nope, D_NOPE, wkc, KV_LORA, q_full, QFULL_W,
              N_TOK, KV_LORA, D_NOPE, 0, true, 8,
              (long)N_TOK * D_NOPE, (long)D_NOPE * KV_LORA, (long)N_TOK * QFULL_W,
              N_HEAD, stream);

  for (int h = 0; h < N_HEAD; ++h) {
    launch_gemm(q_full + (size_t)h * N_TOK * QFULL_W, QFULL_W,
                k_fullT, N_TOK, scores, N_TOK,
                N_TOK, N_TOK, QFULL_W, 1, false, 8, 0, 0, 0, 1, stream);
    softmax_kernel<<<dim3(N_TOK / 8), dim3(256), 0, stream>>>(scores, Pm);
    launch_gemm(Pm, N_TOK, kv_lat, KV_LORA, o_lat, KV_LORA,
                N_TOK, KV_LORA, N_TOK, 2, true, 8, 0, 0, 0, 1, stream);
    launch_gemm(o_lat, KV_LORA, wvc + (size_t)h * KV_LORA * D_VAL, D_VAL,
                o_cat + (size_t)h * D_VAL, N_HEAD * D_VAL,
                N_TOK, D_VAL, KV_LORA, 0, true, 8, 0, 0, 0, 1, stream);
  }

  launch_gemm(o_cat, N_HEAD * D_VAL, wo_b, HID_D, d_out, HID_D,
              N_TOK, HID_D, N_HEAD * D_VAL, 0, false, 8, 0, 0, 0, 1, stream);
}
